// SimpleCoordinatePredictor_16303695855961
// MI455X (gfx1250) — matrix operand order repaired and checked by reading, not compiled, not run on hardware
//
#include <hip/hip_runtime.h>
#include <hip/hip_bf16.h>

typedef __attribute__((ext_vector_type(16))) _Float16 v16h;
typedef __attribute__((ext_vector_type(8)))  _Float16 v8h;
typedef __attribute__((ext_vector_type(2)))  _Float16 v2h;
typedef __attribute__((ext_vector_type(8)))  float    v8f;

#define FEAT 64

// ---------------- degree / normalization ----------------
__global__ void k_fill_deg(float* __restrict__ deg, int N) {
  int i = blockIdx.x * blockDim.x + threadIdx.x;
  if (i < N) deg[i] = 1.0f;  // self loop
}

__global__ void k_count_deg(const int* __restrict__ dst, float* __restrict__ deg, int E) {
  int i = blockIdx.x * blockDim.x + threadIdx.x;
  if (i < E) atomicAdd(&deg[dst[i]], 1.0f);
}

__global__ void k_rsqrt_inplace(float* __restrict__ d, int N) {
  int i = blockIdx.x * blockDim.x + threadIdx.x;
  if (i < N) d[i] = rsqrtf(d[i]);  // deg >= 1 always (self loops)
}

// ---------------- layer-1 pre-feature: g = x * W1 (N x 1 @ 1 x 64) ----------------
__global__ void k_l1pre(const float* __restrict__ x, const float* __restrict__ W1,
                        _Float16* __restrict__ g16, int N) {
  int t = blockIdx.x * blockDim.x + threadIdx.x;
  if (t < N * FEAT) {
    int i = t >> 6, j = t & 63;
    g16[t] = (_Float16)(x[i] * W1[j]);
  }
}

// ---------------- weight fp32 [64][64] row-major -> fp16 transposed [n][k] ----------------
__global__ void k_wconv_t(const float* __restrict__ W, _Float16* __restrict__ Wt) {
  int t = blockIdx.x * blockDim.x + threadIdx.x;
  if (t < FEAT * FEAT) {
    int k = t >> 6, n = t & 63;
    Wt[n * FEAT + k] = (_Float16)W[t];
  }
}

// ---------------- aggregation: self-loop seed ----------------
__global__ void k_agg_init(const _Float16* __restrict__ g16, const float* __restrict__ dinv,
                           float* __restrict__ acc, int N) {
  int t = blockIdx.x * blockDim.x + threadIdx.x;
  if (t < N * FEAT) {
    int i = t >> 6;
    float s = dinv[i];
    acc[t] = (float)g16[t] * (s * s);
  }
}

// ---------------- aggregation: edge scatter-add (4 threads / edge, 16 feats / thread) ----------------
__global__ void k_agg_edge(const int* __restrict__ src, const int* __restrict__ dst,
                           const _Float16* __restrict__ g16, const float* __restrict__ dinv,
                           float* __restrict__ acc, int E) {
  int t = blockIdx.x * blockDim.x + threadIdx.x;
  int e = t >> 2;
  if (e >= E) return;
  int fo = (t & 3) * 16;
  int s = src[e], d = dst[e];
  float w = dinv[s] * dinv[d];
  const _Float16* gp = g16 + (size_t)s * FEAT + fo;
  v8h v0 = *(const v8h*)gp;
  v8h v1 = *(const v8h*)(gp + 8);
  float* ap = acc + (size_t)d * FEAT + fo;
#pragma unroll
  for (int i = 0; i < 8; ++i) atomicAdd(ap + i, (float)v0[i] * w);
#pragma unroll
  for (int i = 0; i < 8; ++i) atomicAdd(ap + 8 + i, (float)v1[i] * w);
}

// ---------------- finalize: h = relu(acc + b) as fp16 ----------------
__global__ void k_finalize(const float* __restrict__ acc, const float* __restrict__ b,
                           _Float16* __restrict__ h16, int N) {
  int t = blockIdx.x * blockDim.x + threadIdx.x;
  if (t < N * FEAT) {
    float v = acc[t] + b[t & 63];
    h16[t] = (_Float16)fmaxf(v, 0.0f);
  }
}

// ---------------- WMMA GEMM: O[N,64] = H[N,64] @ W[64,64] (+bias, relu) ----------------
// Wave-per-row-tile; C tiles staged through LDS (column-major) so global stores are b128.
template <bool BIAS, bool RELU>
__global__ void k_gemm64_wmma(const _Float16* __restrict__ H16,
                              const _Float16* __restrict__ Wt16,  // [n][k] fp16
                              const float* __restrict__ bias,
                              _Float16* __restrict__ O16, int N) {
  __shared__ _Float16 ctile[8][16 * FEAT];  // per-wave 16x64 tile, column-major [n][m]

  const int lane = threadIdx.x & 31;
  const int l16  = lane & 15;
  const int half = lane >> 4;
  const int wib  = threadIdx.x >> 5;  // wave in block
  const int wavesPerBlk = blockDim.x >> 5;
  const int wave   = blockIdx.x * wavesPerBlk + wib;
  const int nwaves = gridDim.x * wavesPerBlk;
  _Float16* my = ctile[wib];

  // B fragments: 2 K-chunks x 4 N-tiles, resident in VGPRs for all row tiles.
  // B layout (contiguous k order, same as A): lane column = lane%16, k = 16*(lane/16) + elem + 32*kc.
  v16h bf[2][4];
#pragma unroll
  for (int kc = 0; kc < 2; ++kc)
#pragma unroll
    for (int nt = 0; nt < 4; ++nt) {
      const _Float16* p = Wt16 + (size_t)(nt * 16 + l16) * FEAT + kc * 32 + half * 16;
      v8h lo = *(const v8h*)p;
      v8h hi = *(const v8h*)(p + 8);
      bf[kc][nt] = __builtin_shufflevector(lo, hi, 0, 1, 2, 3, 4, 5, 6, 7,
                                           8, 9, 10, 11, 12, 13, 14, 15);
    }

  // bias for this lane's 4 output columns, hoisted out of the tile loop
  float bv[4] = {0.f, 0.f, 0.f, 0.f};
  if (BIAS) {
#pragma unroll
    for (int nt = 0; nt < 4; ++nt) bv[nt] = bias[nt * 16 + l16];
  }

  const v8f zero = {0.f, 0.f, 0.f, 0.f, 0.f, 0.f, 0.f, 0.f};
  const int ntiles = (N + 15) >> 4;
  for (int t = wave; t < ntiles; t += nwaves) {
    int arow = t * 16 + l16;  // A layout: lane row = lane%16
    if (arow >= N) arow = N - 1;
    const _Float16* pa = H16 + (size_t)arow * FEAT;

    // Preload BOTH A fragments first so all 4 b128 loads issue as one clause,
    // then run the 8 WMMAs back-to-back.
    // A layout: elem i -> k = 16*(lane/16)+i (two contiguous b128)
    v16h af[2];
#pragma unroll
    for (int kc = 0; kc < 2; ++kc) {
      const _Float16* q = pa + kc * 32 + half * 16;
      v8h alo = *(const v8h*)q;
      v8h ahi = *(const v8h*)(q + 8);
      af[kc] = __builtin_shufflevector(alo, ahi, 0, 1, 2, 3, 4, 5, 6, 7,
                                       8, 9, 10, 11, 12, 13, 14, 15);
    }

    v8f c[4] = {zero, zero, zero, zero};
#pragma unroll
    for (int kc = 0; kc < 2; ++kc) {
      c[0] = __builtin_amdgcn_wmma_f32_16x16x32_f16(false, af[kc], false, bf[kc][0], (short)0, c[0], false, false);
      c[1] = __builtin_amdgcn_wmma_f32_16x16x32_f16(false, af[kc], false, bf[kc][1], (short)0, c[1], false, false);
      c[2] = __builtin_amdgcn_wmma_f32_16x16x32_f16(false, af[kc], false, bf[kc][2], (short)0, c[2], false, false);
      c[3] = __builtin_amdgcn_wmma_f32_16x16x32_f16(false, af[kc], false, bf[kc][3], (short)0, c[3], false, false);
    }

    // Convert + stage into LDS column-major: element (n, m) at my[n*16 + m].
    // C/D layout: VGPR r -> row m = 8*(lane/16)+r, col n = nt*16 + lane%16,
    // so the 8 values of c[nt] are contiguous in m -> one ds_store_b128 each.
#pragma unroll
    for (int nt = 0; nt < 4; ++nt) {
      v8h h;
#pragma unroll
      for (int r = 0; r < 8; ++r) {
        float v = c[nt][r];
        if (BIAS) v += bv[nt];
        if (RELU) v = fmaxf(v, 0.0f);
        h[r] = (_Float16)v;
      }
      *(v8h*)(my + (nt * 16 + l16) * 16 + half * 8) = h;
    }

    // Wave-synchronous readback (DS ops are in-order within a wave).
    // Lane -> row-pair p = lane&7 (rows 2p, 2p+1), column-quarter q = lane>>3.
    // Rows are adjacent in LDS (m stride 1), so one ds_load_b32 yields both rows
    // of a column: 16 b32 LDS loads -> two output rows x 16 cols, 4x b128 stores.
    {
      const int p  = lane & 7;
      const int q  = lane >> 3;
      const int m0 = t * 16 + 2 * p;
      const int m1 = m0 + 1;
      const _Float16* cp = my + 2 * p;
      v8h row0[2], row1[2];
#pragma unroll
      for (int j = 0; j < 2; ++j)
#pragma unroll
        for (int i = 0; i < 8; ++i) {
          v2h w = *(const v2h*)(cp + (q * 16 + j * 8 + i) * 16);
          row0[j][i] = w[0];
          row1[j][i] = w[1];
        }
      if (m0 < N) {
        _Float16* op = O16 + (size_t)m0 * FEAT + q * 16;
        *(v8h*)op = row0[0];
        *(v8h*)(op + 8) = row0[1];
      }
      if (m1 < N) {
        _Float16* op = O16 + (size_t)m1 * FEAT + q * 16;
        *(v8h*)op = row1[0];
        *(v8h*)(op + 8) = row1[1];
      }
    }
  }
}

// ---------------- head: out[N,3] = q[N,64] @ Wp2[64,3] + bp2 ----------------
__global__ void k_head(const _Float16* __restrict__ q16, const float* __restrict__ Wp2,
                       const float* __restrict__ bp2, float* __restrict__ out, int N) {
  int i = blockIdx.x * blockDim.x + threadIdx.x;
  if (i >= N) return;
  const _Float16* r = q16 + (size_t)i * FEAT;
  float a0 = bp2[0], a1 = bp2[1], a2 = bp2[2];
#pragma unroll
  for (int c = 0; c < 8; ++c) {
    v8h v = *(const v8h*)(r + c * 8);
#pragma unroll
    for (int u = 0; u < 8; ++u) {
      int k = c * 8 + u;
      float h = (float)v[u];
      a0 += h * Wp2[k * 3 + 0];
      a1 += h * Wp2[k * 3 + 1];
      a2 += h * Wp2[k * 3 + 2];
    }
  }
  out[(size_t)i * 3 + 0] = a0;
  out[(size_t)i * 3 + 1] = a1;
  out[(size_t)i * 3 + 2] = a2;
}

extern "C" void kernel_launch(void* const* d_in, const int* in_sizes, int n_in,
                              void* d_out, int out_size, void* d_ws, size_t ws_size,
                              hipStream_t stream) {
  const float* x   = (const float*)d_in[0];
  const int*   ei  = (const int*)d_in[1];
  const float* W1  = (const float*)d_in[2];
  const float* b1  = (const float*)d_in[3];
  const float* W2  = (const float*)d_in[4];
  const float* b2  = (const float*)d_in[5];
  const float* W3  = (const float*)d_in[6];
  const float* b3  = (const float*)d_in[7];
  const float* Wp1 = (const float*)d_in[8];
  const float* bp1 = (const float*)d_in[9];
  const float* Wp2 = (const float*)d_in[10];
  const float* bp2 = (const float*)d_in[11];
  (void)n_in; (void)out_size; (void)ws_size;

  const int N = in_sizes[0];          // x is [N,1]
  const int E = in_sizes[1] / 2;      // edge_index is [2,E]
  const int* src = ei;
  const int* dst = ei + E;

  // workspace carve-out (256B aligned)
  char* ws = (char*)d_ws;
  size_t off = 0;
  auto carve = [&](size_t bytes) -> char* {
    char* p = ws + off;
    off += (bytes + 255) & ~(size_t)255;
    return p;
  };
  float*    dinv = (float*)carve((size_t)N * 4);
  _Float16* g16  = (_Float16*)carve((size_t)N * FEAT * 2);  // pre-aggregation features
  float*    acc  = (float*)carve((size_t)N * FEAT * 4);     // aggregation accumulator
  _Float16* h16  = (_Float16*)carve((size_t)N * FEAT * 2);  // layer activations
  _Float16* wt2  = (_Float16*)carve(FEAT * FEAT * 2);
  _Float16* wt3  = (_Float16*)carve(FEAT * FEAT * 2);
  _Float16* wtp  = (_Float16*)carve(FEAT * FEAT * 2);

  const int TB = 256;
  const int gN  = (N + TB - 1) / TB;
  const int gNH = (N * FEAT + TB - 1) / TB;
  const int gE  = (E + TB - 1) / TB;
  const int gE4 = (E * 4 + TB - 1) / TB;
  const int gemmBlk = (((N + 15) / 16) + 7) / 8;  // 8 waves/block, 1 row-tile/wave

  // normalization
  k_fill_deg<<<gN, TB, 0, stream>>>(dinv, N);
  k_count_deg<<<gE, TB, 0, stream>>>(dst, dinv, E);
  k_rsqrt_inplace<<<gN, TB, 0, stream>>>(dinv, N);

  // weight conversions (fp32 -> fp16 transposed)
  k_wconv_t<<<16, TB, 0, stream>>>(W2, wt2);
  k_wconv_t<<<16, TB, 0, stream>>>(W3, wt3);
  k_wconv_t<<<16, TB, 0, stream>>>(Wp1, wtp);

  // ---- layer 1: h = relu(agg(x @ W1) + b1) ----
  k_l1pre<<<gNH, TB, 0, stream>>>(x, W1, g16, N);
  k_agg_init<<<gNH, TB, 0, stream>>>(g16, dinv, acc, N);
  k_agg_edge<<<gE4, TB, 0, stream>>>(src, dst, g16, dinv, acc, E);
  k_finalize<<<gNH, TB, 0, stream>>>(acc, b1, h16, N);

  // ---- layer 2 ----
  k_gemm64_wmma<false, false><<<gemmBlk, TB, 0, stream>>>(h16, wt2, nullptr, g16, N);
  k_agg_init<<<gNH, TB, 0, stream>>>(g16, dinv, acc, N);
  k_agg_edge<<<gE4, TB, 0, stream>>>(src, dst, g16, dinv, acc, E);
  k_finalize<<<gNH, TB, 0, stream>>>(acc, b2, h16, N);

  // ---- layer 3 ----
  k_gemm64_wmma<false, false><<<gemmBlk, TB, 0, stream>>>(h16, wt3, nullptr, g16, N);
  k_agg_init<<<gNH, TB, 0, stream>>>(g16, dinv, acc, N);
  k_agg_edge<<<gE4, TB, 0, stream>>>(src, dst, g16, dinv, acc, E);
  k_finalize<<<gNH, TB, 0, stream>>>(acc, b3, h16, N);

  // ---- head MLP ----
  k_gemm64_wmma<true, true><<<gemmBlk, TB, 0, stream>>>(h16, wtp, bp1, g16, N);
  k_head<<<gN, TB, 0, stream>>>(g16, Wp2, bp2, (float*)d_out, N);
}
